// BinaryTreeLatentVariable_18468359372938
// MI455X (gfx1250) — matrix operand order repaired and checked by reading, not compiled, not run on hardware
//
#include <hip/hip_runtime.h>
#include <cmath>

typedef __bf16 v8bf  __attribute__((ext_vector_type(8)));
typedef __bf16 v16bf __attribute__((ext_vector_type(16)));
typedef float  v8f   __attribute__((ext_vector_type(8)));

#define B_    64
#define NN    1023
#define DIN   512
#define L_    5
#define C_    4
#define Q_    20            // L*C outputs per node
#define K2_   400           // 20*20 flattened (l,r) contraction
#define NCH   13            // ceil(400/32) K-chunks for the level GEMM
#define MTOT  (B_ * NN)     // 65472 rows for the big GEMM
#define HSTR  40            // LDS row stride (bf16) for h tiles: conflict-free b128

// -------------------------------------------------------------------------
// Kernel 0a: W B-frag image, frag order: wf[c][t][lane][e] (bf16)
//   value = W[c*32 + 16*(lane/16) + e][16*t + lane%16], zero-padded cols>=20
// -------------------------------------------------------------------------
__global__ void k_wfrag(const float* __restrict__ W, __bf16* __restrict__ wf) {
    int idx = blockIdx.x * blockDim.x + threadIdx.x;   // 16*2*32*16 = 16384
    if (idx >= 16384) return;
    int e = idx & 15, l = (idx >> 4) & 31, t = (idx >> 9) & 1, c = idx >> 10;
    int row = c * 32 + 16 * (l >> 4) + e;              // < 512
    int col = 16 * t + (l & 15);
    float v = (col < Q_) ? W[row * Q_ + col] : 0.0f;
    wf[idx] = (__bf16)v;
}

// -------------------------------------------------------------------------
// Kernel 0b: Eflat B-frag image: efg[c][t][lane][e] (bf16)
//   Eflat[jk][q] = exp(trans[lp,ll,lr,cp,cl,cr]); jk>=400 or q>=20 -> 0
// -------------------------------------------------------------------------
__global__ void k_efrag(const float* __restrict__ trans, __bf16* __restrict__ efg) {
    int idx = blockIdx.x * blockDim.x + threadIdx.x;   // 13*2*32*16 = 13312
    if (idx >= NCH * 1024) return;
    int e = idx & 15, l = (idx >> 4) & 31, t = (idx >> 9) & 1, c = idx >> 10;
    int jk = c * 32 + 16 * (l >> 4) + e;               // <= 415
    int q  = 16 * t + (l & 15);
    float v = 0.0f;
    if (jk < K2_ && q < Q_) {
        int lp = q / 4, cp = q % 4;
        int j = jk / 20, k = jk % 20;
        int ll = j / 4, cl = j % 4;
        int lr = k / 4, cr = k % 4;
        int tt = ((((lp * L_ + ll) * L_ + lr) * C_ + cp) * C_ + cl) * C_ + cr;
        v = expf(trans[tt]);
    }
    efg[idx] = (__bf16)v;
}

// -------------------------------------------------------------------------
// Kernel 1: sw[row][q] = sum_k h[row][k]*W[k][q] + bias[q]
//   256 threads = 8 waves; block tile = 128 rows; wave tile = 16 rows x 32 cols.
//   h staged to LDS bf16 (stride 40 -> conflict-free b128 frag loads);
//   W fragments read directly from the global frag image (L2-resident).
// -------------------------------------------------------------------------
__global__ void k_gemm_sw(const float* __restrict__ h, const __bf16* __restrict__ wfragG,
                          const float* __restrict__ bias, float* __restrict__ sw) {
    extern __shared__ char smemraw[];
    __bf16* hs = (__bf16*)smemraw;        // 128 x HSTR(40)

    const int tid  = threadIdx.x;
    const int lane = tid & 31, wid = tid >> 5;
    const int m = lane & 15, g = lane >> 4;
    const int blockRow0 = blockIdx.x * 128;

    const v16bf* wf = (const v16bf*)wfragG;
    v8f acc0 = {}; v8f acc1 = {};

    for (int c = 0; c < 16; ++c) {
        const int k0 = c * 32;
        // ---- stage 128x32 chunk of h as bf16 (two threads per row) ----
        {
            int r = tid >> 1, hf = tid & 1;
            int row = blockRow0 + r; if (row >= MTOT) row = MTOT - 1;
            const float* gsrc = h + (size_t)row * DIN + k0 + hf * 16;
            float4 a0 = *(const float4*)(gsrc + 0);
            float4 a1 = *(const float4*)(gsrc + 4);
            float4 a2 = *(const float4*)(gsrc + 8);
            float4 a3 = *(const float4*)(gsrc + 12);
            __bf16* dst = hs + r * HSTR + hf * 16;
            v8bf lo, hi;
            lo[0] = (__bf16)a0.x; lo[1] = (__bf16)a0.y;
            lo[2] = (__bf16)a0.z; lo[3] = (__bf16)a0.w;
            lo[4] = (__bf16)a1.x; lo[5] = (__bf16)a1.y;
            lo[6] = (__bf16)a1.z; lo[7] = (__bf16)a1.w;
            hi[0] = (__bf16)a2.x; hi[1] = (__bf16)a2.y;
            hi[2] = (__bf16)a2.z; hi[3] = (__bf16)a2.w;
            hi[4] = (__bf16)a3.x; hi[5] = (__bf16)a3.y;
            hi[6] = (__bf16)a3.z; hi[7] = (__bf16)a3.w;
            *(v8bf*)(dst + 0) = lo;           // 16B aligned (stride 80B)
            *(v8bf*)(dst + 8) = hi;
            if (k0 + 32 < DIN) __builtin_prefetch(gsrc + 32, 0, 1);  // global_prefetch_b8
        }
        // ---- B fragments: one aligned 32B vector load each (global, L2) ----
        v16bf b0 = wf[(c * 2 + 0) * 32 + lane];
        v16bf b1 = wf[(c * 2 + 1) * 32 + lane];
        __syncthreads();
        // ---- A fragment: two b128 LDS loads + shuffle ----
        const __bf16* arow = hs + (wid * 16 + m) * HSTR;
        v8bf alo = *(const v8bf*)(arow + 16 * g);
        v8bf ahi = *(const v8bf*)(arow + 16 * g + 8);
        v16bf a = __builtin_shufflevector(alo, ahi,
                    0, 1, 2, 3, 4, 5, 6, 7, 8, 9, 10, 11, 12, 13, 14, 15);
        acc0 = __builtin_amdgcn_wmma_f32_16x16x32_bf16(false, a, false, b0,
                                                       (short)0, acc0, false, false);
        acc1 = __builtin_amdgcn_wmma_f32_16x16x32_bf16(false, a, false, b1,
                                                       (short)0, acc1, false, false);
        __syncthreads();
    }

    // ---- epilogue: add bias, store the 20 valid columns ----
#pragma unroll
    for (int v = 0; v < 8; ++v) {
        int rr  = wid * 16 + v + 8 * g;       // C layout: row = v + 8*(lane/16)
        int row = blockRow0 + rr;
        if (row < MTOT) {
            float* o = sw + (size_t)row * Q_;
            o[m] = acc0[v] + bias[m];
            int q1 = 16 + m;
            if (q1 < Q_) o[q1] = acc1[v] + bias[q1];
        }
    }
}

// -------------------------------------------------------------------------
// Kernel 2 (per tree level, P parents): for each (b, node) row
//   out[q] = p[q] + lmax + rmax + log( sum_jk el[j]*er[k]*E[jk][q] )
// GEMM [B*P, 416] x [416, 20] via bf16 WMMA; A frags (el*er) built on the fly;
// Eflat fragments read from the global frag image (L2-resident).
// 128 threads = 4 waves; block tile = 64 rows; wave tile = 16 rows.
// -------------------------------------------------------------------------
__global__ void k_level(const float* __restrict__ childbuf, const float* __restrict__ swbuf,
                        float* __restrict__ inw, const __bf16* __restrict__ efragG,
                        int P, float* __restrict__ outp) {
    extern __shared__ char smemraw[];
    float* elr = (float*)smemraw;    // 64 x 44  [el(20) | er(20) | lmax | rmax | pad]

    const int tid  = threadIdx.x;    // blockDim.x == 128
    const int lane = tid & 31, wid = tid >> 5;
    const int m = lane & 15, g = lane >> 4;
    const int rows = B_ * P;
    const int blk0 = blockIdx.x * 64;

    // per-row setup: 2 threads per row, one child each
    {
        int rr = tid >> 1, side = tid & 1;
        int rho = blk0 + rr;
        if (rho < rows) {
            int b = rho / P, i = rho % P;
            int node  = P - 1 + i;
            int child = 2 * node + 1 + side;
            const float* src = childbuf + ((size_t)b * NN + child) * Q_;
            float x[Q_]; float mx = -3.0e38f;
#pragma unroll
            for (int j = 0; j < Q_; ++j) { x[j] = src[j]; mx = fmaxf(mx, x[j]); }
            float* dst = elr + rr * 44 + side * 20;
#pragma unroll
            for (int j = 0; j < Q_; ++j) dst[j] = expf(x[j] - mx);
            elr[rr * 44 + 40 + side] = mx;
        }
    }
    __syncthreads();

    const v16bf* ef = (const v16bf*)efragG;
    v8f acc0 = {}; v8f acc1 = {};
    const float* me_el = elr + (wid * 16 + m) * 44;   // el
    const float* me_er = me_el + 20;                  // er

    for (int c = 0; c < NCH; ++c) {
        // B fragments: aligned 32B vector loads (global, L2)
        v16bf b0 = ef[(c * 2 + 0) * 32 + lane];
        v16bf b1 = ef[(c * 2 + 1) * 32 + lane];
        // A fragment: batch f32 LDS reads + muls first, convert after
        const int b8 = c * 32 + 16 * g;
        float p0[8], p1[8];
#pragma unroll
        for (int e = 0; e < 8; ++e) {
            int jk = b8 + e;                       // <= 415; E rows >= 400 are zero
            p0[e] = me_el[jk / 20] * me_er[jk % 20];
        }
#pragma unroll
        for (int e = 0; e < 8; ++e) {
            int jk = b8 + 8 + e;
            p1[e] = me_el[jk / 20] * me_er[jk % 20];
        }
        v16bf a;
#pragma unroll
        for (int e = 0; e < 8; ++e) {
            a[e]     = (__bf16)p0[e];
            a[e + 8] = (__bf16)p1[e];
        }
        acc0 = __builtin_amdgcn_wmma_f32_16x16x32_bf16(false, a, false, b0,
                                                       (short)0, acc0, false, false);
        acc1 = __builtin_amdgcn_wmma_f32_16x16x32_bf16(false, a, false, b1,
                                                       (short)0, acc1, false, false);
    }

    // epilogue: log, add p + lmax + rmax, store
#pragma unroll
    for (int v = 0; v < 8; ++v) {
        int rr  = wid * 16 + v + 8 * g;
        int rho = blk0 + rr;
        if (rho < rows) {
            int b = rho / P, i = rho % P;
            int node = P - 1 + i;
            size_t base = ((size_t)b * NN + node) * Q_;
            float mx = elr[rr * 44 + 40] + elr[rr * 44 + 41];
            float v0 = swbuf[base + m] + mx + logf(acc0[v]);
            inw[base + m] = v0;
            if (outp) outp[b * Q_ + m] = v0;       // only level d==0 (node==0)
            int q1 = 16 + m;
            if (q1 < Q_) {
                float v1 = swbuf[base + q1] + mx + logf(acc1[v]);
                inw[base + q1] = v1;
                if (outp) outp[b * Q_ + q1] = v1;
            }
        }
    }
}

// -------------------------------------------------------------------------
extern "C" void kernel_launch(void* const* d_in, const int* in_sizes, int n_in,
                              void* d_out, int out_size, void* d_ws, size_t ws_size,
                              hipStream_t stream) {
    (void)in_sizes; (void)n_in; (void)out_size; (void)ws_size;
    const float* h     = (const float*)d_in[0];  // [64,1023,512]
    const float* W     = (const float*)d_in[1];  // [512,20]
    const float* bias  = (const float*)d_in[2];  // [20]
    const float* trans = (const float*)d_in[3];  // [5,5,5,4,4,4]
    float* out = (float*)d_out;                  // [64,5,4]

    float*  wsf    = (float*)d_ws;
    float*  swbuf  = wsf;                                    // 65472*20 f32
    float*  inw    = wsf + (size_t)MTOT * Q_;                // 65472*20 f32
    __bf16* wfragG = (__bf16*)(wsf + 2 * (size_t)MTOT * Q_); // 16384 bf16 (32B-aligned)
    __bf16* efragG = wfragG + 16384;                         // 13312 bf16

    k_wfrag<<<64, 256, 0, stream>>>(W, wfragG);
    k_efrag<<<(NCH * 1024 + 255) / 256, 256, 0, stream>>>(trans, efragG);

    int g1 = (MTOT + 127) / 128;
    size_t sh1 = (size_t)128 * HSTR * sizeof(__bf16);        // 10.0 KB
    k_gemm_sw<<<g1, 256, sh1, stream>>>(h, wfragG, bias, swbuf);

    size_t sh2 = (size_t)64 * 44 * sizeof(float);            // 11.0 KB
    for (int d = 8; d >= 0; --d) {
        int P = 1 << d;
        int grid = (B_ * P + 63) / 64;
        const float* childbuf = (d == 8) ? swbuf : inw;  // level-8 children are leaves
        float* op = (d == 0) ? out : nullptr;
        k_level<<<grid, 128, sh2, stream>>>(childbuf, swbuf, inw, efragG, P, op);
    }
}
